// TSHMBlockSimple_64106681860635
// MI455X (gfx1250) — matrix operand order repaired and checked by reading, not compiled, not run on hardware
//
#include <hip/hip_runtime.h>

typedef __attribute__((ext_vector_type(16))) _Float16 v16h;
typedef __attribute__((ext_vector_type(8)))  _Float16 v8h;
typedef __attribute__((ext_vector_type(8)))  float    v8f;
typedef __attribute__((ext_vector_type(4)))  unsigned int v4u;
typedef __attribute__((ext_vector_type(4)))  int v4i;
typedef __attribute__((ext_vector_type(8)))  int v8i;

typedef __attribute__((address_space(3))) _Float16 as3_h;
typedef __attribute__((address_space(3))) void     as3_v;
typedef __attribute__((address_space(1))) const void as1_cv;

#if __has_builtin(__builtin_amdgcn_tensor_load_to_lds) && __has_builtin(__builtin_amdgcn_s_wait_tensorcnt)
#define TSH_TDM 1
#elif __has_builtin(__builtin_amdgcn_global_load_async_to_lds_b128) && __has_builtin(__builtin_amdgcn_s_wait_asynccnt)
#define TSH_ASYNC 1
#endif

constexpr int Dm = 1024;   // model dim
constexpr int Rr = 32;     // rank
constexpr int Kk = 8;      // mixing dim
constexpr int FF = 4096;   // ffn dim
constexpr int Bb = 4;      // batch
constexpr int Ll = 4096;   // seq len
constexpr int NT = Bb * Ll;   // 16384 tokens

// ---------------- WMMA helpers (wave32, gfx1250 layouts) ----------------

__device__ inline v8f wmma16(v16h a, v16h b, v8f c) {
  return __builtin_amdgcn_wmma_f32_16x16x32_f16(
      /*neg_a=*/false, a, /*neg_b=*/false, b,
      /*c_mod=*/(short)0, c, /*reuse_a=*/false, /*reuse_b=*/false);
}

// A fragment: 16x32 f16, row m = lane&15, half = lane>>4.
__device__ inline v16h load_a(const _Float16* row_k0, int half) {
  v8h lo = *(const v8h*)(row_k0 + half * 8);
  v8h hi = *(const v8h*)(row_k0 + 16 + half * 8);
  v16h r;
#pragma unroll
  for (int i = 0; i < 8; i++) { r[i] = lo[i]; r[i + 8] = hi[i]; }
  return r;
}

// B fragment: 32x16 f16, col n = lane&15, element i -> k = half*8 + i (i<8), 16 + half*8 + (i-8) (i>=8)
__device__ inline v16h load_b(const _Float16* row_k0, int half) {
  v8h lo = *(const v8h*)(row_k0 + half * 8);
  v8h hi = *(const v8h*)(row_k0 + 16 + half * 8);
  v16h r;
#pragma unroll
  for (int i = 0; i < 8; i++) { r[i] = lo[i]; r[i + 8] = hi[i]; }
  return r;
}

// Ping-pong pipelined K loop: 16xK A strip vs 4 B strips, 4 accumulators.
// ksteps must be even. No buffer-rotation movs: two register sets, peeled tail.
__device__ inline void gemm_k_loop(const _Float16* __restrict__ arow,
                                   const _Float16* __restrict__ b0row,
                                   const _Float16* __restrict__ b1row,
                                   const _Float16* __restrict__ b2row,
                                   const _Float16* __restrict__ b3row,
                                   int ksteps, int half, v8f acc[4]) {
  v16h a0 = load_a(arow, half);
  v16h c00 = load_b(b0row, half);
  v16h c01 = load_b(b1row, half);
  v16h c02 = load_b(b2row, half);
  v16h c03 = load_b(b3row, half);
  v16h a1, c10, c11, c12, c13;
  for (int kt = 0; kt < ksteps - 2; kt += 2) {
    int o1 = (kt + 1) * 32, o2 = (kt + 2) * 32;
    a1  = load_a(arow + o1, half);
    c10 = load_b(b0row + o1, half);
    c11 = load_b(b1row + o1, half);
    c12 = load_b(b2row + o1, half);
    c13 = load_b(b3row + o1, half);
    acc[0] = wmma16(a0, c00, acc[0]);
    acc[1] = wmma16(a0, c01, acc[1]);
    acc[2] = wmma16(a0, c02, acc[2]);
    acc[3] = wmma16(a0, c03, acc[3]);
    a0  = load_a(arow + o2, half);
    c00 = load_b(b0row + o2, half);
    c01 = load_b(b1row + o2, half);
    c02 = load_b(b2row + o2, half);
    c03 = load_b(b3row + o2, half);
    acc[0] = wmma16(a1, c10, acc[0]);
    acc[1] = wmma16(a1, c11, acc[1]);
    acc[2] = wmma16(a1, c12, acc[2]);
    acc[3] = wmma16(a1, c13, acc[3]);
  }
  int o1 = (ksteps - 1) * 32;
  a1  = load_a(arow + o1, half);
  c10 = load_b(b0row + o1, half);
  c11 = load_b(b1row + o1, half);
  c12 = load_b(b2row + o1, half);
  c13 = load_b(b3row + o1, half);
  acc[0] = wmma16(a0, c00, acc[0]);
  acc[1] = wmma16(a0, c01, acc[1]);
  acc[2] = wmma16(a0, c02, acc[2]);
  acc[3] = wmma16(a0, c03, acc[3]);
  acc[0] = wmma16(a1, c10, acc[0]);
  acc[1] = wmma16(a1, c11, acc[1]);
  acc[2] = wmma16(a1, c12, acc[2]);
  acc[3] = wmma16(a1, c13, acc[3]);
}

// ---------------- small utility kernels ----------------

__global__ void cvt_f16_kernel(const float* __restrict__ s, _Float16* __restrict__ d, int n) {
  int i = blockIdx.x * 256 + threadIdx.x;
  if (i < n) d[i] = (_Float16)s[i];
}

__global__ void pack_uv_kernel(const float* __restrict__ U, const float* __restrict__ V,
                               _Float16* __restrict__ uvh, _Float16* __restrict__ uvhT) {
  int i = blockIdx.x * 256 + threadIdx.x;
  if (i >= 64 * Dm) return;
  int r = i / Dm, d = i % Dm;
  float val = (r < 32) ? U[r * Dm + d] : V[(r - 32) * Dm + d];
  _Float16 h = (_Float16)val;
  uvh[i] = h;
  uvhT[d * 64 + r] = h;
}

__device__ inline float block_reduce_256(float v, float* sm) {
  int t = threadIdx.x;
  sm[t] = v; __syncthreads();
  for (int s = 128; s > 0; s >>= 1) { if (t < s) sm[t] += sm[t + s]; __syncthreads(); }
  float r = sm[0]; __syncthreads();
  return r;
}

// ---------------- LN1 + tanh -> phi (f16) ; gate conv partial dots ----------------

__global__ __launch_bounds__(256) void ln1_phi_gate(
    const float* __restrict__ X, const float* __restrict__ g, const float* __restrict__ bta,
    const float* __restrict__ gw, _Float16* __restrict__ phih, float* __restrict__ gdot) {
  __shared__ float sm[256];
  int n = blockIdx.x, t = threadIdx.x;
  const float* x = X + (size_t)n * Dm;
  float v[4]; float s = 0.f, ss = 0.f;
#pragma unroll
  for (int i = 0; i < 4; i++) { float xv = x[t + 256 * i]; v[i] = xv; s += xv; ss += xv * xv; }
  s = block_reduce_256(s, sm); ss = block_reduce_256(ss, sm);
  float mean = s * (1.f / Dm);
  float var = ss * (1.f / Dm) - mean * mean;
  float inv = rsqrtf(var + 1e-5f);
  float t0 = 0.f, t1 = 0.f, t2 = 0.f;
#pragma unroll
  for (int i = 0; i < 4; i++) {
    int d = t + 256 * i;
    float xn = (v[i] - mean) * inv * g[d] + bta[d];
    phih[(size_t)n * Dm + d] = (_Float16)tanhf(xn);
    t0 += xn * gw[d * 3 + 0];
    t1 += xn * gw[d * 3 + 1];
    t2 += xn * gw[d * 3 + 2];
  }
  t0 = block_reduce_256(t0, sm);
  t1 = block_reduce_256(t1, sm);
  t2 = block_reduce_256(t2, sm);
  if (t == 0) { gdot[n * 3 + 0] = t0; gdot[n * 3 + 1] = t1; gdot[n * 3 + 2] = t2; }
}

// ---------------- P,Q = phi @ [U|V]^T  (WMMA, N=64) ----------------

__global__ __launch_bounds__(256) void pq_gemm(
    const _Float16* __restrict__ phih, const _Float16* __restrict__ uvh,
    float* __restrict__ P, float* __restrict__ Q) {
  int wave = threadIdx.x >> 5, lane = threadIdx.x & 31;
  int half = lane >> 4, l15 = lane & 15;
  int m0 = (blockIdx.x * 8 + wave) * 16;
  const _Float16* arow = phih + (size_t)(m0 + l15) * Dm;
  v8f acc[4] = {};
  gemm_k_loop(arow,
              uvh + (size_t)(0 * 16 + l15) * Dm,
              uvh + (size_t)(1 * 16 + l15) * Dm,
              uvh + (size_t)(2 * 16 + l15) * Dm,
              uvh + (size_t)(3 * 16 + l15) * Dm,
              Dm / 32, half, acc);
#pragma unroll
  for (int nt = 0; nt < 4; nt++) {
    int n = nt * 16 + l15;
#pragma unroll
    for (int v = 0; v < 8; v++) {
      int mm = m0 + v + half * 8;
      float val = acc[nt][v];
      if (n < 32) P[(size_t)mm * Rr + n] = val;
      else        Q[(size_t)mm * Rr + (n - 32)] = val;
    }
  }
}

// ---------------- prefix / suffix cumsum over L (per batch, lane = r) ----------------

__global__ void cumsum_fwd(const float* __restrict__ P, float* __restrict__ S) {
  int b = blockIdx.x, r = threadIdx.x;
  size_t base = (size_t)b * Ll * Rr + r;
  float acc = 0.f;
  for (int l = 0; l < Ll; l++) { acc += P[base + (size_t)l * Rr]; S[base + (size_t)l * Rr] = acc; }
}

__global__ void cumsum_bwd(const float* __restrict__ M, float* __restrict__ Ms) {
  int b = blockIdx.x, r = threadIdx.x;
  size_t base = (size_t)b * Ll * Rr + r;
  float acc = 0.f;
  for (int l = Ll - 1; l >= 0; l--) { acc += M[base + (size_t)l * Rr]; Ms[base + (size_t)l * Rr] = acc; }
}

// ---------------- emission: e,G and M = G@A_w, N = G@B_w  (wave per token) ----------------

__global__ __launch_bounds__(256) void emission(
    const float* __restrict__ S, const float* __restrict__ Qv,
    const float* __restrict__ A_w, const float* __restrict__ A_b,
    const float* __restrict__ B_w, const float* __restrict__ B_b,
    const float* __restrict__ c, float* __restrict__ Mv, float* __restrict__ Nv) {
  int wave = threadIdx.x >> 5, lane = threadIdx.x & 31;
  int n = blockIdx.x * 8 + wave;
  float s = S[(size_t)n * Rr + lane];
  float q = Qv[(size_t)n * Rr + lane];
  float m = 0.f, nn = 0.f;
#pragma unroll
  for (int k = 0; k < Kk; k++) {
    float aw = A_w[k * Rr + lane], bw = B_w[k * Rr + lane];
    float p = s * aw + q * bw;
#pragma unroll
    for (int off = 16; off > 0; off >>= 1) p += __shfl_xor(p, off, 32);
    float gk = 0.001f * tanhf(p + A_b[k] + B_b[k] + c[k]);   // 0.1 * 0.1^2
    m += gk * aw;
    nn += gk * bw;
  }
  Mv[(size_t)n * Rr + lane] = m;
  Nv[(size_t)n * Rr + lane] = nn;
}

// ---------------- grad = [Msuf|N] @ [U;V] (WMMA K=64) + gate + residual ----------------

__global__ __launch_bounds__(256) void grad_gate(
    const float* __restrict__ X, const float* __restrict__ Ms, const float* __restrict__ Nv,
    const _Float16* __restrict__ uvhT, const float* __restrict__ gdot,
    const float* __restrict__ gate_b, const float* __restrict__ res_scale,
    float* __restrict__ Xnext) {
  int wave = threadIdx.x >> 5, lane = threadIdx.x & 31;
  int half = lane >> 4, l15 = lane & 15;
  int m0 = blockIdx.x * 16;
  int d0base = (blockIdx.y * 8 + wave) * 64;
  int m = m0 + l15;
  v16h a0, a1;
#pragma unroll
  for (int i = 0; i < 16; i++) {
    int kl = (i < 8) ? (half * 8 + i) : (16 + half * 8 + (i - 8));
    a0[i] = (_Float16)Ms[(size_t)m * Rr + kl];
    a1[i] = (_Float16)Nv[(size_t)m * Rr + kl];
  }
  int l = m & (Ll - 1);
  float t1 = gdot[m * 3 + 1];
  float t0 = (l > 0) ? gdot[(m - 1) * 3 + 0] : 0.f;
  float t2 = (l < Ll - 1) ? gdot[(m + 1) * 3 + 2] : 0.f;
  float graw = t0 + t1 + t2 + gate_b[0];
  float gval = 1e-6f + (1.f - 1e-6f) / (1.f + expf(-graw));
  float rs = res_scale[0];
#pragma unroll
  for (int nt = 0; nt < 4; nt++) {
    int d0 = d0base + nt * 16;
    const _Float16* brow = uvhT + (size_t)(d0 + l15) * 64;
    v8f acc = {};
    acc = wmma16(a0, load_b(brow, half), acc);
    acc = wmma16(a1, load_b(brow + 32, half), acc);
#pragma unroll
    for (int v = 0; v < 8; v++) {
      int ml = v + half * 8;
      float gm = __shfl(gval, ml, 32);
      size_t idx = (size_t)(m0 + ml) * Dm + d0 + l15;
      Xnext[idx] = X[idx] + rs * gm * acc[v];
    }
  }
}

// ---------------- LN2 -> h_in (f16) ----------------

__global__ __launch_bounds__(256) void ln2_kernel(
    const float* __restrict__ Xn, const float* __restrict__ g, const float* __restrict__ bta,
    _Float16* __restrict__ hin) {
  __shared__ float sm[256];
  int n = blockIdx.x, t = threadIdx.x;
  const float* x = Xn + (size_t)n * Dm;
  float v[4]; float s = 0.f, ss = 0.f;
#pragma unroll
  for (int i = 0; i < 4; i++) { float xv = x[t + 256 * i]; v[i] = xv; s += xv; ss += xv * xv; }
  s = block_reduce_256(s, sm); ss = block_reduce_256(ss, sm);
  float mean = s * (1.f / Dm);
  float var = ss * (1.f / Dm) - mean * mean;
  float inv = rsqrtf(var + 1e-5f);
#pragma unroll
  for (int i = 0; i < 4; i++) {
    int d = t + 256 * i;
    hin[(size_t)n * Dm + d] = (_Float16)((v[i] - mean) * inv * g[d] + bta[d]);
  }
}

// ---------------- FFN GEMM1: block stages A tile via TDM, waves sweep FF ----------------

__global__ __launch_bounds__(256) void ffn1_gemm(
    const _Float16* __restrict__ hin, const _Float16* __restrict__ w1h,
    const float* __restrict__ b1, _Float16* __restrict__ hmid) {
  __shared__ _Float16 As[16 * Dm];   // 32 KB: 16 tokens x D, shared by all 8 waves
  int wave = threadIdx.x >> 5, lane = threadIdx.x & 31;
  int half = lane >> 4, l15 = lane & 15;
  int m0 = blockIdx.x * 16;

#if defined(TSH_TDM)
  // Tensor Data Mover: DMA the 16 x 1024 f16 tile into LDS (one wave issues).
  if (threadIdx.x < 32) {
    unsigned ldsOff = (unsigned)(size_t)(as3_h*)As;
    unsigned long long ga = (unsigned long long)(const void*)(hin + (size_t)m0 * Dm);
    v4u g0;
    g0[0] = 1u;                                               // count=1, user mode
    g0[1] = ldsOff;                                           // lds_addr (bytes)
    g0[2] = (unsigned)(ga & 0xFFFFFFFFull);                   // global_addr[31:0]
    g0[3] = (unsigned)((ga >> 32) & 0x1FFFFFFull) | (2u << 30); // addr[56:32] | type=2
    v8i g1;
    g1[0] = (int)(1u << 16);                                  // data_size=1 -> 2 bytes
    g1[1] = (int)((unsigned)Dm << 16);                        // tensor_dim0[15:0]
    g1[2] = (int)(((unsigned)Dm >> 16) | (((unsigned)NT & 0xFFFFu) << 16)); // dim0 hi | dim1 lo
    g1[3] = (int)((((unsigned)NT >> 16) & 0xFFFFu) | ((unsigned)Dm << 16)); // dim1 hi | tile_dim0
    g1[4] = 16;                                               // tile_dim1 = 16 rows
    g1[5] = Dm;                                               // tensor_dim0_stride[31:0]
    g1[6] = 0;
    g1[7] = 0;
    v4i zz = {0, 0, 0, 0};
#if defined(__clang_major__) && __clang_major__ >= 23
    v8i z8 = {0, 0, 0, 0, 0, 0, 0, 0};
    __builtin_amdgcn_tensor_load_to_lds(g0, g1, zz, zz, z8, 0);
#else
    __builtin_amdgcn_tensor_load_to_lds(g0, g1, zz, zz, 0);
#endif
    __builtin_amdgcn_s_wait_tensorcnt(0);
  }
  __syncthreads();
#elif defined(TSH_ASYNC)
  {
    const char* src = (const char*)(hin + (size_t)m0 * Dm);
    char* dst = (char*)As;
    for (int i = threadIdx.x; i < (16 * Dm) / 8; i += 256) {
      __builtin_amdgcn_global_load_async_to_lds_b128(
          (as1_cv*)(src + (size_t)i * 16), (as3_v*)(dst + (size_t)i * 16), 0, 0);
    }
    __builtin_amdgcn_s_wait_asynccnt(0);
    __syncthreads();
  }
#else
  {
    const v8h* src = (const v8h*)(hin + (size_t)m0 * Dm);
    v8h* dst = (v8h*)As;
    for (int i = threadIdx.x; i < (16 * Dm) / 8; i += 256) dst[i] = src[i];
    __syncthreads();
  }
#endif

  const _Float16* arow = As + (size_t)l15 * Dm;
  for (int ff = 0; ff < 8; ff++) {
    int f0 = (ff * 8 + wave) * 64;
    v8f acc[4] = {};
    gemm_k_loop(arow,
                w1h + (size_t)(f0 + 0 * 16 + l15) * Dm,
                w1h + (size_t)(f0 + 1 * 16 + l15) * Dm,
                w1h + (size_t)(f0 + 2 * 16 + l15) * Dm,
                w1h + (size_t)(f0 + 3 * 16 + l15) * Dm,
                Dm / 32, half, acc);
#pragma unroll
    for (int nt = 0; nt < 4; nt++) {
      int f = f0 + nt * 16 + l15;
      float bias = b1[f];
#pragma unroll
      for (int v = 0; v < 8; v++) {
        int mm = m0 + v + half * 8;
        float x = acc[nt][v] + bias;
        float ge = 0.5f * x * (1.f + erff(x * 0.70710678118f));
        hmid[(size_t)mm * FF + f] = (_Float16)ge;
      }
    }
  }
}

// ---------------- FFN GEMM2: out += Hmid @ w2^T + b2  (WMMA) ----------------

__global__ __launch_bounds__(256) void ffn2_gemm(
    const _Float16* __restrict__ hmid, const _Float16* __restrict__ w2h,
    const float* __restrict__ b2, float* __restrict__ out) {
  int wave = threadIdx.x >> 5, lane = threadIdx.x & 31;
  int half = lane >> 4, l15 = lane & 15;
  int m0 = blockIdx.x * 16;
  int d0 = (blockIdx.y * 8 + wave) * 64;
  const _Float16* arow = hmid + (size_t)(m0 + l15) * FF;
  v8f acc[4] = {};
  gemm_k_loop(arow,
              w2h + (size_t)(d0 + 0 * 16 + l15) * FF,
              w2h + (size_t)(d0 + 1 * 16 + l15) * FF,
              w2h + (size_t)(d0 + 2 * 16 + l15) * FF,
              w2h + (size_t)(d0 + 3 * 16 + l15) * FF,
              FF / 32, half, acc);
#pragma unroll
  for (int nt = 0; nt < 4; nt++) {
    int dd = d0 + nt * 16 + l15;
    float bias = b2[dd];
#pragma unroll
    for (int v = 0; v < 8; v++) {
      size_t idx = (size_t)(m0 + v + half * 8) * Dm + dd;
      out[idx] = out[idx] + acc[nt][v] + bias;
    }
  }
}

// ---------------- host launch ----------------

extern "C" void kernel_launch(void* const* d_in, const int* in_sizes, int n_in,
                              void* d_out, int out_size, void* d_ws, size_t ws_size,
                              hipStream_t stream) {
  const float* X      = (const float*)d_in[0];
  const float* U_w    = (const float*)d_in[1];
  const float* V_w    = (const float*)d_in[2];
  const float* A_w    = (const float*)d_in[3];
  const float* A_b    = (const float*)d_in[4];
  const float* B_w    = (const float*)d_in[5];
  const float* B_b    = (const float*)d_in[6];
  const float* c      = (const float*)d_in[7];
  const float* gate_w = (const float*)d_in[8];
  const float* gate_b = (const float*)d_in[9];
  const float* res_s  = (const float*)d_in[10];
  const float* ln1_g  = (const float*)d_in[11];
  const float* ln1_b  = (const float*)d_in[12];
  const float* ln2_g  = (const float*)d_in[13];
  const float* ln2_b  = (const float*)d_in[14];
  const float* w1     = (const float*)d_in[15];
  const float* b1     = (const float*)d_in[16];
  const float* w2     = (const float*)d_in[17];
  const float* b2     = (const float*)d_in[18];
  float* out = (float*)d_out;

  char* p = (char*)d_ws;
  auto carve = [&](size_t bytes) -> void* {
    void* r = (void*)p;
    p += (bytes + 255) & ~(size_t)255;
    return r;
  };
  _Float16* w1h  = (_Float16*)carve((size_t)FF * Dm * sizeof(_Float16));
  _Float16* w2h  = (_Float16*)carve((size_t)Dm * FF * sizeof(_Float16));
  _Float16* uvh  = (_Float16*)carve((size_t)64 * Dm * sizeof(_Float16));
  _Float16* uvhT = (_Float16*)carve((size_t)Dm * 64 * sizeof(_Float16));
  _Float16* phih = (_Float16*)carve((size_t)NT * Dm * sizeof(_Float16));
  float*    gdot = (float*)carve((size_t)NT * 3 * sizeof(float));
  float*    Pb   = (float*)carve((size_t)NT * Rr * sizeof(float));
  float*    Qb   = (float*)carve((size_t)NT * Rr * sizeof(float));
  float*    Sb   = (float*)carve((size_t)NT * Rr * sizeof(float));
  float*    Mv   = (float*)carve((size_t)NT * Rr * sizeof(float));
  float*    Msuf = (float*)carve((size_t)NT * Rr * sizeof(float));
  float*    Nv   = (float*)carve((size_t)NT * Rr * sizeof(float));
  _Float16* hin  = (_Float16*)carve((size_t)NT * Dm * sizeof(_Float16));
  _Float16* hmid = (_Float16*)carve((size_t)NT * FF * sizeof(_Float16));

  { int n = FF * Dm; cvt_f16_kernel<<<(n + 255) / 256, 256, 0, stream>>>(w1, w1h, n); }
  { int n = Dm * FF; cvt_f16_kernel<<<(n + 255) / 256, 256, 0, stream>>>(w2, w2h, n); }
  { int n = 64 * Dm; pack_uv_kernel<<<(n + 255) / 256, 256, 0, stream>>>(U_w, V_w, uvh, uvhT); }

  ln1_phi_gate<<<NT, 256, 0, stream>>>(X, ln1_g, ln1_b, gate_w, phih, gdot);
  pq_gemm<<<NT / 16 / 8, 256, 0, stream>>>(phih, uvh, Pb, Qb);
  cumsum_fwd<<<Bb, 32, 0, stream>>>(Pb, Sb);
  emission<<<NT / 8, 256, 0, stream>>>(Sb, Qb, A_w, A_b, B_w, B_b, c, Mv, Nv);
  cumsum_bwd<<<Bb, 32, 0, stream>>>(Mv, Msuf);
  grad_gate<<<dim3(NT / 16, 2), 256, 0, stream>>>(X, Msuf, Nv, uvhT, gdot, gate_b, res_s, out);
  ln2_kernel<<<NT, 256, 0, stream>>>(out, ln2_g, ln2_b, hin);
  ffn1_gemm<<<NT / 16, 256, 0, stream>>>(hin, w1h, b1, hmid);
  ffn2_gemm<<<dim3(NT / 16, 2), 256, 0, stream>>>(hmid, w2h, b2, out);

  (void)in_sizes; (void)n_in; (void)out_size; (void)ws_size;
}
